// DomainAttentionLayer_16578573763134
// MI455X (gfx1250) — matrix operand order repaired and checked by reading, not compiled, not run on hardware
//
#include <hip/hip_runtime.h>

// ---------------------------------------------------------------------------
// DomainAttentionLayer on MI455X (gfx1250), wave32 + WMMA + async-LDS staging.
//   q = x@Wq^T+bq ; k = dx@Wk^T+bk ; v = dx@Wv^T+bv
//   out = softmax(q k^T / sqrt(D)) v        N=M=8192, D=512
// Pipeline: f32->f16 convert -> 3x WMMA projection GEMMs (v stored transposed)
//        -> split-M flash attention (k/v tiles async-staged to LDS, double
//           buffered; streaming softmax) -> merge kernel.
// ---------------------------------------------------------------------------

#define DDIM 512
#define NROW 8192
#define MROW 8192
#define SPLITM 4
#define MCHUNK (MROW / SPLITM)     // 2048 keys per split

#define KPAD 520                   // 512 + 8 halves: conflict-free k rows
#define VPAD 40                    // 32 + 8 halves: conflict-free v^T rows

typedef __attribute__((ext_vector_type(16))) _Float16 v16h;
typedef __attribute__((ext_vector_type(8)))  float    v8f;
typedef int v4i __attribute__((vector_size(16)));
typedef __attribute__((address_space(1))) v4i* gas_v4i_ptr;   // global (AS1)
typedef __attribute__((address_space(3))) v4i* las_v4i_ptr;   // LDS (AS3)

union F16Frag { v16h v; uint4 q[2]; };   // 16 halves = one WMMA A/B fragment
union F16x8   { _Float16 h[8]; uint4 q; };

// -------- CDNA5 async global->LDS copy (ASYNCcnt-tracked), 16B per lane -----
__device__ __forceinline__ void async_copy16(void* lds_dst, const void* gsrc) {
#if __has_builtin(__builtin_amdgcn_global_load_async_to_lds_b128)
  __builtin_amdgcn_global_load_async_to_lds_b128(
      (gas_v4i_ptr)(void*)gsrc, (las_v4i_ptr)lds_dst, 0, 0);
#else
  unsigned int loff = (unsigned int)(unsigned long long)lds_dst; // LDS offset
  asm volatile("global_load_async_to_lds_b128 %0, %1, off"
               :: "v"(loff), "v"(gsrc) : "memory");
#endif
}

__device__ __forceinline__ void wait_async_le16() {
#if __has_builtin(__builtin_amdgcn_s_wait_asynccnt)
  __builtin_amdgcn_s_wait_asynccnt(16);
#else
  asm volatile("s_wait_asynccnt 0x10" ::: "memory");
#endif
}

// ------------------------- f32 -> f16 convert ------------------------------
__global__ void cvt_f32_f16(const float* __restrict__ in,
                            _Float16* __restrict__ out, int n) {
  int i = blockIdx.x * blockDim.x + threadIdx.x;
  int stride = gridDim.x * blockDim.x;
  for (; i < n; i += stride) out[i] = (_Float16)in[i];
}

// ------------------------- projection GEMM ---------------------------------
// out[i,j] = sum_d A[i,d]*W[j,d] + bias[j].  8 waves: wave w -> rows
// [blk.x*128+16w,+16), cols [blk.y*64,+64) as 4 WMMA col-tiles, K-steps of 32.
__global__ __launch_bounds__(256)
void proj_wmma(const _Float16* __restrict__ A,   // [rows,512] f16
               const _Float16* __restrict__ W,   // [512,512]  f16 [j,d]
               const float*    __restrict__ bias,// [512]
               _Float16*       __restrict__ out, // [rows,512] or [512,rows]
               int rowsTotal, int transposeOut)
{
  const int lane   = threadIdx.x & 31;
  const int wave   = threadIdx.x >> 5;
  const int laneHi = lane >> 4;
  const int ln     = lane & 15;
  const int hi8    = laneHi * 8;
  const int hi16   = laneHi * 16;
  const int rowBase = blockIdx.x * 128 + wave * 16;
  const int colBase = blockIdx.y * 64;

  v8f acc[4] = {};
  const _Float16* arow = A + (size_t)(rowBase + ln) * DDIM;

  for (int kk = 0; kk < DDIM; kk += 32) {
    F16Frag a;
    a.q[0] = *(const uint4*)(arow + kk + hi8);
    a.q[1] = *(const uint4*)(arow + kk + 16 + hi8);
#pragma unroll
    for (int t = 0; t < 4; ++t) {
      const _Float16* wrow = W + (size_t)(colBase + t*16 + ln) * DDIM + kk + hi8;
      F16Frag b;
      b.q[0] = *(const uint4*)(wrow);
      b.q[1] = *(const uint4*)(wrow + 16);
      acc[t] = __builtin_amdgcn_wmma_f32_16x16x32_f16(
          false, a.v, false, b.v, (short)0, acc[t], false, false);
    }
  }

#pragma unroll
  for (int t = 0; t < 4; ++t) {
    const int col = colBase + t*16 + ln;
    const float bv = bias[col];
    if (!transposeOut) {
#pragma unroll
      for (int v = 0; v < 8; ++v) {
        const int row = rowBase + hi8 + v;
        out[(size_t)row * DDIM + col] = (_Float16)(acc[t][v] + bv);
      }
    } else {
      F16x8 pk;
#pragma unroll
      for (int v = 0; v < 8; ++v) pk.h[v] = (_Float16)(acc[t][v] + bv);
      *(uint4*)(out + (size_t)col * rowsTotal + rowBase + hi8) = pk.q;
    }
  }
}

// ------------------------- flash attention (split-M) -----------------------
__device__ __forceinline__ void stage_tiles(
    _Float16* klds, _Float16* vlds, int mb, int tid,
    const _Float16* __restrict__ Kt, const _Float16* __restrict__ VT)
{
  // k tile: 32 rows x 512 halves -> klds rows padded to KPAD. 2048 16B-chunks.
#pragma unroll
  for (int r = 0; r < 8; ++r) {
    const int chunk = r * 256 + tid;
    const int row = chunk >> 6, col = (chunk & 63) << 3;
    async_copy16(klds + row * KPAD + col,
                 Kt + (size_t)(mb + row) * DDIM + col);
  }
  // v^T tile: 512 rows x 32 halves -> vlds rows padded to VPAD.
#pragma unroll
  for (int r = 0; r < 8; ++r) {
    const int chunk = r * 256 + tid;
    const int d = chunk >> 2, mcol = (chunk & 3) << 3;
    async_copy16(vlds + d * VPAD + mcol,
                 VT + (size_t)d * MROW + mb + mcol);
  }
}

__global__ __launch_bounds__(256)
void attn_wmma(const _Float16* __restrict__ Q,   // [N,512]
               const _Float16* __restrict__ Kt,  // [M,512]
               const _Float16* __restrict__ VT,  // [512,M]
               float* __restrict__ opart,        // [SPLITM][N][512]
               float* __restrict__ mpart,        // [SPLITM][N]
               float* __restrict__ lpart)        // [SPLITM][N]
{
  __shared__ __align__(16) _Float16 klds[2][32 * KPAD];    // 2 x 33.3 KB
  __shared__ __align__(16) _Float16 vlds[2][DDIM * VPAD];  // 2 x 40 KB
  __shared__ __align__(16) _Float16 plds[8][16 * 32];      // 8 KB

  const int tid    = threadIdx.x;
  const int lane   = tid & 31;
  const int wave   = tid >> 5;
  const int laneHi = lane >> 4;
  const int ln     = lane & 15;
  const int hi8    = laneHi * 8;
  const int hi16   = laneHi * 16;
  const int qRow   = blockIdx.x * 128 + wave * 16;
  const int split  = blockIdx.y;
  const int mstart = split * MCHUNK;
  const int mend   = mstart + MCHUNK;

  // q A-fragments: 16 K-tiles covering D=512, pinned in VGPRs
  F16Frag qa[16];
  const _Float16* qrow = Q + (size_t)(qRow + ln) * DDIM;
#pragma unroll
  for (int t = 0; t < 16; ++t) {
    qa[t].q[0] = *(const uint4*)(qrow + t*32 + hi8);
    qa[t].q[1] = *(const uint4*)(qrow + t*32 + 16 + hi8);
  }

  v8f oc[32] = {};
  float mrow[8], lrow[8];
#pragma unroll
  for (int v = 0; v < 8; ++v) { mrow[v] = -3.0e38f; lrow[v] = 0.0f; }

  const float scale = 0.044194173824159216f;   // 1/sqrt(512)

  stage_tiles(klds[0], vlds[0], mstart, tid, Kt, VT);   // prefetch first tile

  for (int mb = mstart; mb < mend; mb += 32) {
    const int buf = ((mb - mstart) >> 5) & 1;
    const int nmb = (mb + 32 < mend) ? (mb + 32) : mstart;
    stage_tiles(klds[buf ^ 1], vlds[buf ^ 1], nmb, tid, Kt, VT); // 16 async
    wait_async_le16();            // previous buffer's 16 copies done
    __syncthreads();              // ... for every thread

    // ---- scores s = q @ k^T for 32 keys (2 col-tiles) from LDS ----
    v8f sc[2] = {};
#pragma unroll
    for (int t = 0; t < 16; ++t) {
#pragma unroll
      for (int ct = 0; ct < 2; ++ct) {
        const _Float16* kp = &klds[buf][(ct*16 + ln) * KPAD + t*32 + hi8];
        F16Frag b;
        b.q[0] = *(const uint4*)(kp);
        b.q[1] = *(const uint4*)(kp + 16);
        sc[ct] = __builtin_amdgcn_wmma_f32_16x16x32_f16(
            false, qa[t].v, false, b.v, (short)0, sc[ct], false, false);
      }
    }

    // ---- online softmax per row (row = v + hi8; 16 lanes share a row) ----
#pragma unroll
    for (int v = 0; v < 8; ++v) {
      float s0 = sc[0][v] * scale;
      float s1 = sc[1][v] * scale;
      float rm = fmaxf(s0, s1);
#pragma unroll
      for (int off = 1; off < 16; off <<= 1)
        rm = fmaxf(rm, __shfl_xor(rm, off, 32));
      const float mnew = fmaxf(mrow[v], rm);
      const float corr = __expf(mrow[v] - mnew);
      const float p0 = __expf(s0 - mnew);
      const float p1 = __expf(s1 - mnew);
      float rs = p0 + p1;
#pragma unroll
      for (int off = 1; off < 16; off <<= 1)
        rs += __shfl_xor(rs, off, 32);
      lrow[v] = lrow[v] * corr + rs;
      mrow[v] = mnew;
#pragma unroll
      for (int f = 0; f < 32; ++f) oc[f][v] *= corr;
      plds[wave][(v + hi8)*32 + ln]      = (_Float16)p0;
      plds[wave][(v + hi8)*32 + 16 + ln] = (_Float16)p1;
    }
    asm volatile("s_wait_dscnt 0" ::: "memory");   // wave-local LDS RAW
    F16Frag pa;                                    // p as 16x32 A-fragment
    pa.q[0] = *(const uint4*)(&plds[wave][ln*32 + hi8]);
    pa.q[1] = *(const uint4*)(&plds[wave][ln*32 + 16 + hi8]);

    // ---- o += p @ v from LDS v^T tile ----
#pragma unroll
    for (int f = 0; f < 32; ++f) {
      const _Float16* vp = &vlds[buf][(f*16 + ln) * VPAD + hi8];
      F16Frag b;
      b.q[0] = *(const uint4*)(vp);
      b.q[1] = *(const uint4*)(vp + 16);
      oc[f] = __builtin_amdgcn_wmma_f32_16x16x32_f16(
          false, pa.v, false, b.v, (short)0, oc[f], false, false);
    }
    __syncthreads();              // all reads of buf done before restage
  }

  // ---- epilogue: store raw partials (o, m, l) ----
#pragma unroll
  for (int v = 0; v < 8; ++v) {
    const int row = qRow + hi8 + v;
    const size_t rowoff = ((size_t)split * NROW + row) * DDIM;
#pragma unroll
    for (int f = 0; f < 32; ++f) opart[rowoff + f*16 + ln] = oc[f][v];
    if (ln == 0) {
      mpart[(size_t)split * NROW + row] = mrow[v];
      lpart[(size_t)split * NROW + row] = lrow[v];
    }
  }
}

// ------------------------- split-M merge -----------------------------------
__global__ void attn_merge(const float* __restrict__ opart,
                           const float* __restrict__ mpart,
                           const float* __restrict__ lpart,
                           float* __restrict__ out, int total) {
  int idx = blockIdx.x * blockDim.x + threadIdx.x;
  const int stride = gridDim.x * blockDim.x;
  for (; idx < total; idx += stride) {
    const int row = idx >> 9;                  // / DDIM
    float m = -3.0e38f;
#pragma unroll
    for (int s = 0; s < SPLITM; ++s) m = fmaxf(m, mpart[s * NROW + row]);
    float num = 0.0f, den = 0.0f;
#pragma unroll
    for (int s = 0; s < SPLITM; ++s) {
      const float w = __expf(mpart[s * NROW + row] - m);
      num += w * opart[(size_t)s * NROW * DDIM + idx];
      den += w * lpart[s * NROW + row];
    }
    out[idx] = num / den;
  }
}

// --------------------------------------------------------------------------
extern "C" void kernel_launch(void* const* d_in, const int* in_sizes, int n_in,
                              void* d_out, int out_size, void* d_ws, size_t ws_size,
                              hipStream_t stream) {
  const float* x  = (const float*)d_in[0];
  const float* dx = (const float*)d_in[1];
  const float* Wq = (const float*)d_in[2];
  const float* bq = (const float*)d_in[3];
  const float* Wk = (const float*)d_in[4];
  const float* bk = (const float*)d_in[5];
  const float* Wv = (const float*)d_in[6];
  const float* bv = (const float*)d_in[7];
  float* out = (float*)d_out;

  // workspace carve-up: f16 buffers (~41.5 MB) + split-M partials (~64.3 MB)
  _Float16* xh  = (_Float16*)d_ws;
  _Float16* dxh = xh  + (size_t)NROW * DDIM;
  _Float16* wqh = dxh + (size_t)MROW * DDIM;
  _Float16* wkh = wqh + (size_t)DDIM * DDIM;
  _Float16* wvh = wkh + (size_t)DDIM * DDIM;
  _Float16* qh  = wvh + (size_t)DDIM * DDIM;
  _Float16* kh  = qh  + (size_t)NROW * DDIM;
  _Float16* vth = kh  + (size_t)MROW * DDIM;
  float* opart  = (float*)(vth + (size_t)MROW * DDIM);
  float* mpart  = opart + (size_t)SPLITM * NROW * DDIM;
  float* lpart  = mpart + (size_t)SPLITM * NROW;

  const int ND = NROW * DDIM, MD = MROW * DDIM, DD = DDIM * DDIM;
  cvt_f32_f16<<<1024, 256, 0, stream>>>(x,  xh,  ND);
  cvt_f32_f16<<<1024, 256, 0, stream>>>(dx, dxh, MD);
  cvt_f32_f16<<<256,  256, 0, stream>>>(Wq, wqh, DD);
  cvt_f32_f16<<<256,  256, 0, stream>>>(Wk, wkh, DD);
  cvt_f32_f16<<<256,  256, 0, stream>>>(Wv, wvh, DD);

  proj_wmma<<<dim3(NROW/128, DDIM/64), 256, 0, stream>>>(xh,  wqh, bq, qh,  NROW, 0);
  proj_wmma<<<dim3(MROW/128, DDIM/64), 256, 0, stream>>>(dxh, wkh, bk, kh,  MROW, 0);
  proj_wmma<<<dim3(MROW/128, DDIM/64), 256, 0, stream>>>(dxh, wvh, bv, vth, MROW, 1);

  attn_wmma<<<dim3(NROW/128, SPLITM), 256, 0, stream>>>(qh, kh, vth,
                                                        opart, mpart, lpart);
  attn_merge<<<4096, 256, 0, stream>>>(opart, mpart, lpart, out, NROW * DDIM);
}
